// WindowAttention_1812476199067
// MI455X (gfx1250) — matrix operand order repaired and checked by reading, not compiled, not run on hardware
//
#include <hip/hip_runtime.h>
#include <hip/hip_bf16.h>

// ---------------------------------------------------------------------------
// Fused Swin window-attention for gfx1250 (MI455X), bf16 WMMA, wave32.
// Memory-bound (~400MB mandatory HBM traffic @23.3TB/s); everything between
// the x read and the out write lives in LDS (133KB/block, 2 blocks/WGP).
// All LDS fragment traffic is 16B-aligned b128; scores/AV are computed
// transposed so softmax needs a single lane^16 shuffle and every D-tile
// write-back packs 8 bf16 into one ds_store_b128. All 8 waves are busy in
// every phase (attention work is chunked across waves, not per-head).
// ---------------------------------------------------------------------------

typedef __bf16 bf16_t;
typedef __attribute__((ext_vector_type(16))) __bf16 v16bf;
typedef __attribute__((ext_vector_type(8)))  float  v8f;
typedef __attribute__((ext_vector_type(4)))  unsigned int u32x4;

union Frag { v16bf v; u32x4 q[2]; };
union Pack { bf16_t h[8]; u32x4 q; };

#define CDIM   192
#define HEADS  6
#define HDIM   32
#define NTOK   64

// LDS strides in bf16 elements (multiples of 8 -> 16B alignment everywhere)
#define XS   200           // x tile : [tok][ch]        64 x 200
#define QKS  392           // q,k    : [tok][o<384]     64 x 392 (q pre-scaled)
#define VTS  72            // v^T    : [c 0..191][tok]  192 x 72
#define ATS  72            // attn   : per head [q][m]  64 x 72
#define AOS  200           // attn@V : [tok][ch]        64 x 200

#define R1_ELEMS (HEADS*64*ATS)      // 27648 (x tile aliases low part)
#define QK_ELEMS (64*QKS)            // 25088 (attn@V output aliases this)
#define VT_ELEMS (CDIM*VTS)          // 13824
#define SMEM_ELEMS (R1_ELEMS + QK_ELEMS + VT_ELEMS)   // 66560 el = 133120 B

#define NWQ   (36*6*32*16)   // qkv_w A-fragments
#define NWO   (12*6*32*16)   // out_w A-fragments
#define NBIAS (HEADS*64*64)  // rpe bias, C-fragment order (transposed scores)
#define NQB   (36*32*8)      // qkv_b, C-fragment order
#define NOB   (12*32*8)      // out_b, C-fragment order

static __device__ __forceinline__ v8f wmma_bf16(v16bf a, v16bf b, v8f c) {
  return __builtin_amdgcn_wmma_f32_16x16x32_bf16(false, a, false, b, (short)0, c,
                                                 false, false);
}

// ---------------------------------------------------------------------------
// Prep: swizzle weights into per-lane A-fragment order; pack all additive
// terms (rpe bias, qkv_b, out_b) into per-lane C-fragment order so the main
// kernel feeds them straight into the WMMA C operand with one v8f load.
// A-frag 16-bit 16x32 layout: element p -> K = ((p<8)?p:p+8) + 8*(lane>>4).
// C-frag layout: element r -> row = r + 8*(lane>>4), col = lane&15.
// ---------------------------------------------------------------------------
__global__ void swin_prep(const float* __restrict__ qkv_w,
                          const float* __restrict__ out_w,
                          const float* __restrict__ qkv_b,
                          const float* __restrict__ out_b,
                          const float* __restrict__ rpe,
                          bf16_t* __restrict__ wq,
                          bf16_t* __restrict__ wo,
                          float*  __restrict__ bias6,
                          float*  __restrict__ qbf,
                          float*  __restrict__ obf) {
  int idx = blockIdx.x * 256 + threadIdx.x;
  if (idx < NWQ) {
    int p = idx & 15, lane = (idx >> 4) & 31, kk = (idx >> 9) % 6, mt = idx / 3072;
    int row = mt * 16 + (lane & 15);
    int col = kk * 32 + ((p < 8) ? p : (p + 8)) + 8 * (lane >> 4);
    wq[idx] = (bf16_t)qkv_w[row * CDIM + col];
  } else if (idx < NWQ + NWO) {
    int i2 = idx - NWQ;
    int p = i2 & 15, lane = (i2 >> 4) & 31, kk = (i2 >> 9) % 6, mt = i2 / 3072;
    int row = mt * 16 + (lane & 15);
    int col = kk * 32 + ((p < 8) ? p : (p + 8)) + 8 * (lane >> 4);
    wo[i2] = (bf16_t)out_w[row * CDIM + col];
  } else if (idx < NWQ + NWO + NBIAS) {
    // scores are computed transposed: D row = k-token, D col = q-token.
    int i3 = idx - NWQ - NWO;
    int r = i3 & 7, lane = (i3 >> 3) & 31;
    int mt = (i3 >> 8) & 3, nt = (i3 >> 10) & 3, h = i3 >> 12;
    int q = nt * 16 + (lane & 15);
    int k = mt * 16 + r + 8 * (lane >> 4);
    int relh = (k >> 3) - (q >> 3) + 7;
    int relw = (k & 7) - (q & 7) + 7;
    bias6[i3] = rpe[h * 225 + relh * 15 + relw];
  } else if (idx < NWQ + NWO + NBIAS + NQB) {
    int i4 = idx - NWQ - NWO - NBIAS;
    int r = i4 & 7, mt = i4 >> 8, hh = (i4 >> 7) & 1;
    qbf[i4] = qkv_b[mt * 16 + r + 8 * hh];
  } else if (idx < NWQ + NWO + NBIAS + NQB + NOB) {
    int i5 = idx - NWQ - NWO - NBIAS - NQB;
    int r = i5 & 7, mt = i5 >> 8, hh = (i5 >> 7) & 1;
    obf[i5] = out_b[mt * 16 + r + 8 * hh];
  }
}

// ---------------------------------------------------------------------------
// Main fused kernel: one 8x8 window per block (256 threads = 8 wave32s).
// ---------------------------------------------------------------------------
__global__ __launch_bounds__(256) void swin_attn(
    const float* __restrict__ x,
    const bf16_t* __restrict__ wq,
    const bf16_t* __restrict__ wo,
    const float* __restrict__ bias6,
    const float* __restrict__ qbf,
    const float* __restrict__ obf,
    float* __restrict__ out) {
  __shared__ __align__(16) bf16_t smem[SMEM_ELEMS];
  bf16_t* xs = smem;                         // phase 1-2 (dead after QKV)
  bf16_t* at = smem;                         // phase 3-4 (aliases xs)
  bf16_t* qk = smem + R1_ELEMS;              // phase 2-3
  bf16_t* ao = qk;                           // phase 4-5 (aliases qk)
  bf16_t* vt = smem + R1_ELEMS + QK_ELEMS;   // phase 2-4

  const int tid  = threadIdx.x;
  const int wave = __builtin_amdgcn_readfirstlane(tid >> 5);  // wave-uniform
  const int lane = tid & 31;
  const int hh   = lane >> 4;     // half-wave select (C/D row offset 8*hh)
  const int ln   = lane & 15;

  const int win = blockIdx.x;          // 4096 = B(4) * hd(32) * wd(32)
  const int b   = win >> 10;
  const int rem = win & 1023;
  const int h0  = (rem >> 5) << 3;
  const int w0  = (rem & 31) << 3;

  // ---- Phase 1: x tile -> LDS token-major bf16 (packed b128 stores) -------
  const size_t xbase = ((size_t)b * CDIM) * 65536 + (size_t)h0 * 256 + w0;
  for (int g = tid; g < NTOK * (CDIM / 8); g += 256) {
    int tok = g & 63, cg = g >> 6;                       // 8-channel group
    const float* xp = x + xbase + (size_t)(cg * 8) * 65536 +
                      (size_t)(tok >> 3) * 256 + (tok & 7);
    Pack p;
    #pragma unroll
    for (int j = 0; j < 8; ++j) p.h[j] = (bf16_t)xp[(size_t)j * 65536];
    *(u32x4*)(xs + tok * XS + cg * 8) = p.q;
  }
  __syncthreads();

  // ---- Phase 2: qkv = Wq @ x + b  (576x64, 144 tiles, 18 per wave) --------
  const float qscale = 0.17677669529663687f;   // 1/sqrt(32), folded into Q
  for (int t = 0; t < 18; ++t) {
    int tile = wave * 18 + t;                  // wave-uniform (scalar)
    int mt = tile >> 2, nt = tile & 3;
    v8f acc = *(const v8f*)(qbf + (mt * 32 + lane) * 8);
    #pragma unroll
    for (int kk = 0; kk < 6; ++kk) {
      const bf16_t* ap = wq + ((size_t)(mt * 6 + kk) * 32 + lane) * 16;
      Frag a; a.q[0] = *(const u32x4*)ap;       a.q[1] = *(const u32x4*)(ap + 8);
      const bf16_t* bp = xs + (nt * 16 + ln) * XS + kk * 32 + 8 * hh;
      Frag bb; bb.q[0] = *(const u32x4*)bp;     bb.q[1] = *(const u32x4*)(bp + 16);
      acc = wmma_bf16(a.v, bb.v, acc);
    }
    int tok = nt * 16 + ln;
    if (mt < 24) {                       // whole tile is Q or K (scalar branch)
      float s = (mt < 12) ? qscale : 1.0f;
      Pack p;
      #pragma unroll
      for (int r = 0; r < 8; ++r) p.h[r] = (bf16_t)(acc[r] * s);
      *(u32x4*)(qk + tok * QKS + mt * 16 + 8 * hh) = p.q;   // 8 consecutive o
    } else {                             // whole tile is V -> channel-major
      #pragma unroll
      for (int r = 0; r < 8; ++r) {
        int c = mt * 16 + r + 8 * hh - 2 * CDIM;
        vt[c * VTS + tok] = (bf16_t)acc[r];
      }
    }
  }
  __syncthreads();

  // ---- Phase 3: scores^T = K Q^T + bias^T, softmax -> attn[q][m] ----------
  // D row = k-token (in-lane r + mt tiles), D col = q-token (lane). Softmax
  // axis = 32 in-lane values + partner half-wave (lane^16). 24 independent
  // (head, q-tile) chunks spread over all 8 waves.
  for (int i = 0; i < 3; ++i) {
    int chunk = wave * 3 + i;                  // wave-uniform
    int h = chunk >> 2, nt = chunk & 3;
    bf16_t* ah = at + h * 64 * ATS;
    const bf16_t* qp = qk + (nt * 16 + ln) * QKS + h * HDIM + 8 * hh;
    Frag qb; qb.q[0] = *(const u32x4*)qp;  qb.q[1] = *(const u32x4*)(qp + 16);
    v8f sc[4];
    #pragma unroll
    for (int mt = 0; mt < 4; ++mt) {           // k tile
      const bf16_t* kp = qk + (mt * 16 + ln) * QKS + CDIM + h * HDIM + 8 * hh;
      Frag ka; ka.q[0] = *(const u32x4*)kp; ka.q[1] = *(const u32x4*)(kp + 16);
      v8f c = *(const v8f*)(bias6 + (((h * 4 + nt) * 4 + mt) * 32 + lane) * 8);
      sc[mt] = wmma_bf16(ka.v, qb.v, c);
    }
    float rm = sc[0][0];
    #pragma unroll
    for (int mt = 0; mt < 4; ++mt)
      #pragma unroll
      for (int r = 0; r < 8; ++r) rm = fmaxf(rm, sc[mt][r]);
    rm = fmaxf(rm, __shfl_xor(rm, 16, 32));
    float rs = 0.f;
    #pragma unroll
    for (int mt = 0; mt < 4; ++mt)
      #pragma unroll
      for (int r = 0; r < 8; ++r) {
        sc[mt][r] = __expf(sc[mt][r] - rm);
        rs += sc[mt][r];
      }
    rs += __shfl_xor(rs, 16, 32);
    float inv = __builtin_amdgcn_rcpf(rs);
    #pragma unroll
    for (int mt = 0; mt < 4; ++mt) {
      Pack p;
      #pragma unroll
      for (int r = 0; r < 8; ++r) p.h[r] = (bf16_t)(sc[mt][r] * inv);
      *(u32x4*)(ah + (nt * 16 + ln) * ATS + mt * 16 + 8 * hh) = p.q;
    }
  }
  __syncthreads();   // all waves done reading qk before ao overwrites it

  // ---- Phase 4: (attn @ V)^T = V^T attn^T -> ao[tok][ch] ------------------
  // A = V^T (c x m) from vt, B = attn^T (m x q) read row-major from ah,
  // D rows = channels -> per-lane r spans 8 consecutive channels => b128 pack.
  // 48 tiles spread over all 8 waves (6 each).
  for (int i = 0; i < 6; ++i) {
    int tile = wave * 6 + i;                   // wave-uniform
    int h = tile >> 3, ntq = (tile >> 1) & 3, ct = tile & 1;
    bf16_t* ah = at + h * 64 * ATS;
    v8f acc = {0.f, 0.f, 0.f, 0.f, 0.f, 0.f, 0.f, 0.f};
    #pragma unroll
    for (int ch = 0; ch < 2; ++ch) {           // K = 64 tokens, 2x32
      const bf16_t* apn = vt + (h * HDIM + ct * 16 + ln) * VTS + ch * 32 + 8 * hh;
      Frag a;  a.q[0] = *(const u32x4*)apn;  a.q[1] = *(const u32x4*)(apn + 16);
      const bf16_t* bpn = ah + (ntq * 16 + ln) * ATS + ch * 32 + 8 * hh;
      Frag bb; bb.q[0] = *(const u32x4*)bpn; bb.q[1] = *(const u32x4*)(bpn + 16);
      acc = wmma_bf16(a.v, bb.v, acc);
    }
    Pack p;
    #pragma unroll
    for (int r = 0; r < 8; ++r) p.h[r] = (bf16_t)acc[r];
    *(u32x4*)(ao + (ntq * 16 + ln) * AOS + h * HDIM + ct * 16 + 8 * hh) = p.q;
  }
  __syncthreads();

  // ---- Phase 5: out = Wo @ ao + out_b, store f32 (48 tiles, 6 per wave) ---
  for (int t = 0; t < 6; ++t) {
    int tile = wave * 6 + t;                   // wave-uniform
    int mt = tile >> 2, nt = tile & 3;
    v8f acc = *(const v8f*)(obf + (mt * 32 + lane) * 8);
    #pragma unroll
    for (int kk = 0; kk < 6; ++kk) {
      const bf16_t* ap = wo + ((size_t)(mt * 6 + kk) * 32 + lane) * 16;
      Frag a;  a.q[0] = *(const u32x4*)ap;   a.q[1] = *(const u32x4*)(ap + 8);
      const bf16_t* bp = ao + (nt * 16 + ln) * AOS + kk * 32 + 8 * hh;
      Frag bb; bb.q[0] = *(const u32x4*)bp;  bb.q[1] = *(const u32x4*)(bp + 16);
      acc = wmma_bf16(a.v, bb.v, acc);
    }
    #pragma unroll
    for (int r = 0; r < 8; ++r) {
      int o   = mt * 16 + r + 8 * hh;
      int tok = nt * 16 + ln;
      out[((size_t)b * CDIM + o) * 65536 +
          (size_t)(h0 + (tok >> 3)) * 256 + (w0 + (tok & 7))] = acc[r];
    }
  }
}

// ---------------------------------------------------------------------------
extern "C" void kernel_launch(void* const* d_in, const int* in_sizes, int n_in,
                              void* d_out, int out_size, void* d_ws, size_t ws_size,
                              hipStream_t stream) {
  const float* x      = (const float*)d_in[0];
  const float* qkv_w  = (const float*)d_in[1];
  const float* qkv_b  = (const float*)d_in[2];
  const float* out_w  = (const float*)d_in[3];
  const float* out_b  = (const float*)d_in[4];
  const float* rpe    = (const float*)d_in[5];
  float* out = (float*)d_out;

  // Workspace: wq | wo | bias6(C-frag) | qkv_b(C-frag) | out_b(C-frag) ~432KB
  char* ws = (char*)d_ws;
  bf16_t* wq    = (bf16_t*)(ws);
  bf16_t* wo    = (bf16_t*)(ws + (size_t)NWQ * 2);
  float*  bias6 = (float*)(ws + (size_t)NWQ * 2 + (size_t)NWO * 2);
  float*  qbf   = (float*)((char*)bias6 + (size_t)NBIAS * 4);
  float*  obf   = (float*)((char*)qbf + (size_t)NQB * 4);

  const int prep_total = NWQ + NWO + NBIAS + NQB + NOB;
  swin_prep<<<(prep_total + 255) / 256, 256, 0, stream>>>(
      qkv_w, out_w, qkv_b, out_b, rpe, wq, wo, bias6, qbf, obf);
  swin_attn<<<4096, 256, 0, stream>>>(x, wq, wo, bias6, qbf, obf, out);
}
